// GRUMemoryWithSentimentModel_34385508171804
// MI455X (gfx1250) — compile-verified, matrix operand order checked by reading, not run on hardware
//
#include <hip/hip_runtime.h>
#include <hip/hip_bf16.h>

// ---------------------------------------------------------------------------
// GRU memory model, MI455X (gfx1250, wave32, WMMA).
// GEMMs use v_wmma_f32_16x16x32_bf16 with LDS-staged bf16 tiles (f32 accum).
// ---------------------------------------------------------------------------

typedef __attribute__((ext_vector_type(16))) __bf16 v16bf;
typedef __attribute__((ext_vector_type(8)))  __bf16 v8bf;
typedef __attribute__((ext_vector_type(8)))  float  v8f;

#define B_BATCH 16384
#define D_HID   768
#define GRU_IN  771
#define H_MEM   128
#define N3H     384     // 3*H
#define NSLOT   65536

#define BM 128
#define BN 128
#define BK 32

// ---------------------------------------------------------------------------
// C[M x N] = A[M x K] * W[N x K]^T   (row-major A, row-major W; f32 in/out,
// bf16 WMMA inside).  Grid: (M/BM, N/BN), block 256 threads = 8 waves (4x2).
// Each wave owns a 32x64 sub-tile = 2x4 wmma accumulators.
// ---------------------------------------------------------------------------
__global__ __launch_bounds__(256)
void wmma_gemm_nt(const float* __restrict__ A, int lda,
                  const float* __restrict__ W, int ldw,
                  float* __restrict__ C, int ldc, int K)
{
    __shared__ alignas(16) __bf16 As[BM * BK];   // 8 KB
    __shared__ alignas(16) __bf16 Ws[BN * BK];   // 8 KB

    const int tid   = threadIdx.x;
    const int lane  = tid & 31;
    const int wave  = tid >> 5;
    const int waveM = wave >> 1;      // 0..3 -> 32-row slab
    const int waveN = wave & 1;       // 0..1 -> 64-col slab
    const int lr    = lane & 15;
    const int lhi   = lane >> 4;      // 0/1
    const int rowBase = blockIdx.x * BM;
    const int colBase = blockIdx.y * BN;

    // staging assignment: each thread stages 16 elements of A and 16 of W
    const int ldRow = tid >> 1;           // 0..127
    const int ldK   = (tid & 1) * 16;     // 0 or 16

    v8f acc[2][4] = {};

    for (int k0 = 0; k0 < K; k0 += BK) {
        // ---- stage A tile: 4x float4 global loads, cvt to bf16 ----
        {
            const float* pa = A + (size_t)(rowBase + ldRow) * lda + k0 + ldK;
            float4 x0 = *(const float4*)(pa + 0);
            float4 x1 = *(const float4*)(pa + 4);
            float4 x2 = *(const float4*)(pa + 8);
            float4 x3 = *(const float4*)(pa + 12);
            v8bf a, b;
            a[0]=(__bf16)x0.x; a[1]=(__bf16)x0.y; a[2]=(__bf16)x0.z; a[3]=(__bf16)x0.w;
            a[4]=(__bf16)x1.x; a[5]=(__bf16)x1.y; a[6]=(__bf16)x1.z; a[7]=(__bf16)x1.w;
            b[0]=(__bf16)x2.x; b[1]=(__bf16)x2.y; b[2]=(__bf16)x2.z; b[3]=(__bf16)x2.w;
            b[4]=(__bf16)x3.x; b[5]=(__bf16)x3.y; b[6]=(__bf16)x3.z; b[7]=(__bf16)x3.w;
            __bf16* d = &As[ldRow * BK + ldK];
            *(v8bf*)(d)     = a;
            *(v8bf*)(d + 8) = b;
        }
        // ---- stage W tile: scalar loads (ldw may be odd, e.g. 771) ----
        {
            const float* pw = W + (size_t)(colBase + ldRow) * ldw + k0 + ldK;
            v8bf a, b;
#pragma unroll
            for (int q = 0; q < 8; ++q) a[q] = (__bf16)pw[q];
#pragma unroll
            for (int q = 0; q < 8; ++q) b[q] = (__bf16)pw[8 + q];
            __bf16* d = &Ws[ldRow * BK + ldK];
            *(v8bf*)(d)     = a;
            *(v8bf*)(d + 8) = b;
        }
        __syncthreads();

        // ISA 16-bit A/B fragment layout: lanes 0-15 hold K {0..7,16..23},
        // lanes 16-31 hold K {8..15,24..31} of row/col (lane & 15).
        const int kA = lhi ? 8 : 0;
        const int kB = lhi ? 24 : 16;

        v16bf af[2];
#pragma unroll
        for (int mf = 0; mf < 2; ++mf) {
            const __bf16* p = &As[(waveM * 32 + mf * 16 + lr) * BK];
            v8bf lo = *(const v8bf*)(p + kA);
            v8bf hi = *(const v8bf*)(p + kB);
            af[mf] = __builtin_shufflevector(lo, hi, 0,1,2,3,4,5,6,7,8,9,10,11,12,13,14,15);
        }
        v16bf bfr[4];
#pragma unroll
        for (int nf = 0; nf < 4; ++nf) {
            const __bf16* p = &Ws[(waveN * 64 + nf * 16 + lr) * BK];
            v8bf lo = *(const v8bf*)(p + kA);
            v8bf hi = *(const v8bf*)(p + kB);
            bfr[nf] = __builtin_shufflevector(lo, hi, 0,1,2,3,4,5,6,7,8,9,10,11,12,13,14,15);
        }
#pragma unroll
        for (int mf = 0; mf < 2; ++mf)
#pragma unroll
            for (int nf = 0; nf < 4; ++nf)
                acc[mf][nf] = __builtin_amdgcn_wmma_f32_16x16x32_bf16(
                    false, af[mf], false, bfr[nf], (short)0, acc[mf][nf], false, false);
        __syncthreads();
    }

    // C fragment layout: (vgpr r, lane L) -> row = r + (L>=16 ? 8:0), col = L&15
#pragma unroll
    for (int mf = 0; mf < 2; ++mf) {
#pragma unroll
        for (int nf = 0; nf < 4; ++nf) {
            const int col = colBase + waveN * 64 + nf * 16 + lr;
#pragma unroll
            for (int r = 0; r < 8; ++r) {
                const int row = rowBase + waveM * 32 + mf * 16 + lhi * 8 + r;
                C[(size_t)row * ldc + col] = acc[mf][nf][r];
            }
        }
    }
}

// ---------------------------------------------------------------------------
// gather: s_t[i,:] = memory[slot_ids[i],:]   (float4 per thread)
// ---------------------------------------------------------------------------
__global__ void gather_rows(const float* __restrict__ memory,
                            const int* __restrict__ slot,
                            float* __restrict__ s_t)
{
    int idx = blockIdx.x * blockDim.x + threadIdx.x;   // i*32 + q
    if (idx >= B_BATCH * (H_MEM / 4)) return;
    int i = idx >> 5, q = idx & 31;
    const float4* src = (const float4*)(memory + (size_t)slot[i] * H_MEM) + q;
    ((float4*)(s_t + (size_t)i * H_MEM))[q] = *src;
}

// ---------------------------------------------------------------------------
// GRU gate math; folds biases + rank-3 sentiment update into the epilogue.
// ---------------------------------------------------------------------------
__global__ void gru_gates(const float* __restrict__ gi, const float* __restrict__ gh,
                          const float* __restrict__ s_t, const float* __restrict__ sent,
                          const float* __restrict__ W_ih,
                          const float* __restrict__ b_ih, const float* __restrict__ b_hh,
                          float* __restrict__ s_new)
{
    int idx = blockIdx.x * blockDim.x + threadIdx.x;   // i*H + j
    if (idx >= B_BATCH * H_MEM) return;
    int i = idx >> 7, j = idx & 127;

    float s0 = sent[i * 3 + 0], s1 = sent[i * 3 + 1], s2 = sent[i * 3 + 2];
    const float* girow = gi + (size_t)i * N3H;
    const float* ghrow = gh + (size_t)i * N3H;

    // sentiment columns 768..770 of W_ih (not covered by the K=768 GEMM)
    const float* w0 = W_ih + (size_t)(j)       * GRU_IN + D_HID;
    const float* w1 = W_ih + (size_t)(j + 128) * GRU_IN + D_HID;
    const float* w2 = W_ih + (size_t)(j + 256) * GRU_IN + D_HID;

    float ir = girow[j]       + b_ih[j]       + s0*w0[0] + s1*w0[1] + s2*w0[2];
    float iz = girow[j + 128] + b_ih[j + 128] + s0*w1[0] + s1*w1[1] + s2*w1[2];
    float in_= girow[j + 256] + b_ih[j + 256] + s0*w2[0] + s1*w2[1] + s2*w2[2];
    float hr = ghrow[j]       + b_hh[j];
    float hz = ghrow[j + 128] + b_hh[j + 128];
    float hn = ghrow[j + 256] + b_hh[j + 256];

    float r = 1.0f / (1.0f + __expf(-(ir + hr)));
    float z = 1.0f / (1.0f + __expf(-(iz + hz)));
    float n = tanhf(in_ + r * hn);
    s_new[idx] = (1.0f - z) * n + z * s_t[idx];
}

// ---------------------------------------------------------------------------
// heads: one wave per row; 8 dot products over z_t = [h_t, sent, s_t, h_ctx].
// ---------------------------------------------------------------------------
__global__ void heads_kernel(const float* __restrict__ h_t, const float* __restrict__ sent,
                             const float* __restrict__ s_t, const float* __restrict__ h_ctx,
                             const float* __restrict__ W_e, const float* __restrict__ b_e,
                             const float* __restrict__ W_s, const float* __restrict__ b_s,
                             float* __restrict__ out_e, float* __restrict__ out_s)
{
    int gwave = (blockIdx.x * blockDim.x + threadIdx.x) >> 5;
    int lane  = threadIdx.x & 31;
    if (gwave >= B_BATCH) return;
    const int i = gwave;

    float acc[8] = {0.f,0.f,0.f,0.f,0.f,0.f,0.f,0.f};

    // segment 1: h_t -> fusion cols [0, 768)
    for (int c = lane; c < D_HID; c += 32) {
        float v = h_t[(size_t)i * D_HID + c];
#pragma unroll
        for (int j = 0; j < 7; ++j) acc[j] += v * W_e[(size_t)j * 1667 + c];
        acc[7] += v * W_s[c];
    }
    // segment 2: sentiment -> cols [768, 771)
    if (lane < 3) {
        float v = sent[i * 3 + lane];
#pragma unroll
        for (int j = 0; j < 7; ++j) acc[j] += v * W_e[(size_t)j * 1667 + 768 + lane];
        acc[7] += v * W_s[768 + lane];
    }
    // segment 3: s_t -> cols [771, 899)
    for (int c = lane; c < H_MEM; c += 32) {
        float v = s_t[(size_t)i * H_MEM + c];
#pragma unroll
        for (int j = 0; j < 7; ++j) acc[j] += v * W_e[(size_t)j * 1667 + 771 + c];
        acc[7] += v * W_s[771 + c];
    }
    // segment 4: h_ctx -> cols [899, 1667)
    for (int c = lane; c < D_HID; c += 32) {
        float v = h_ctx[(size_t)i * D_HID + c];
#pragma unroll
        for (int j = 0; j < 7; ++j) acc[j] += v * W_e[(size_t)j * 1667 + 899 + c];
        acc[7] += v * W_s[899 + c];
    }

#pragma unroll
    for (int j = 0; j < 8; ++j) {
        float a = acc[j];
        for (int off = 16; off > 0; off >>= 1) a += __shfl_xor(a, off, 32);
        acc[j] = a;
    }
    if (lane == 0) {
#pragma unroll
        for (int j = 0; j < 7; ++j) out_e[(size_t)i * 7 + j] = acc[j] + b_e[j];
        out_s[i] = acc[7] + b_s[0];
    }
}

// ---------------------------------------------------------------------------
// deterministic last-write-wins scatter
// ---------------------------------------------------------------------------
__global__ void winner_init(int* __restrict__ w)
{
    int s = blockIdx.x * blockDim.x + threadIdx.x;
    if (s < NSLOT) w[s] = -1;
}
__global__ void winner_mark(const int* __restrict__ slot, int* __restrict__ w)
{
    int i = blockIdx.x * blockDim.x + threadIdx.x;
    if (i < B_BATCH) atomicMax(&w[slot[i]], i);
}
__global__ void scatter_rows(const int* __restrict__ slot, const int* __restrict__ w,
                             const float* __restrict__ s_new, float* __restrict__ out_mem)
{
    int idx = blockIdx.x * blockDim.x + threadIdx.x;   // i*32 + q
    if (idx >= B_BATCH * (H_MEM / 4)) return;
    int i = idx >> 5, q = idx & 31;
    int s = slot[i];
    if (w[s] == i)
        ((float4*)(out_mem + (size_t)s * H_MEM))[q] =
            ((const float4*)(s_new + (size_t)i * H_MEM))[q];
}

// ---------------------------------------------------------------------------
extern "C" void kernel_launch(void* const* d_in, const int* in_sizes, int n_in,
                              void* d_out, int out_size, void* d_ws, size_t ws_size,
                              hipStream_t stream)
{
    const float* h_t    = (const float*)d_in[0];
    const float* h_ctx  = (const float*)d_in[1];
    const float* sent   = (const float*)d_in[2];
    const int*   slot   = (const int*)d_in[3];
    const float* memory = (const float*)d_in[4];
    const float* W_ih   = (const float*)d_in[5];
    const float* W_hh   = (const float*)d_in[6];
    const float* b_ih   = (const float*)d_in[7];
    const float* b_hh   = (const float*)d_in[8];
    const float* W_e    = (const float*)d_in[9];
    const float* b_e    = (const float*)d_in[10];
    const float* W_s    = (const float*)d_in[11];
    const float* b_s    = (const float*)d_in[12];

    float* out_e   = (float*)d_out;
    float* out_s   = out_e + (size_t)B_BATCH * 7;
    float* out_mem = out_s + B_BATCH;

    char*  ws  = (char*)d_ws;
    size_t off = 0;
    float* s_t    = (float*)(ws + off); off += (size_t)B_BATCH * H_MEM * sizeof(float);
    float* s_new  = (float*)(ws + off); off += (size_t)B_BATCH * H_MEM * sizeof(float);
    float* gi     = (float*)(ws + off); off += (size_t)B_BATCH * N3H  * sizeof(float);
    float* gh     = (float*)(ws + off); off += (size_t)B_BATCH * N3H  * sizeof(float);
    int*   winner = (int*)  (ws + off); off += (size_t)NSLOT * sizeof(int);
    (void)ws_size; (void)in_sizes; (void)n_in; (void)out_size;

    // 1) gather pre-update memory rows
    gather_rows<<<(B_BATCH * 32 + 255) / 256, 256, 0, stream>>>(memory, slot, s_t);

    // 2) gi = h_t @ W_ih[:, :768]^T  (sentiment cols folded into gates)
    wmma_gemm_nt<<<dim3(B_BATCH / BM, N3H / BN), 256, 0, stream>>>(
        h_t, D_HID, W_ih, GRU_IN, gi, N3H, D_HID);

    // 3) gh = s_t @ W_hh^T
    wmma_gemm_nt<<<dim3(B_BATCH / BM, N3H / BN), 256, 0, stream>>>(
        s_t, H_MEM, W_hh, H_MEM, gh, N3H, H_MEM);

    // 4) gates -> s_new
    gru_gates<<<(B_BATCH * H_MEM + 255) / 256, 256, 0, stream>>>(
        gi, gh, s_t, sent, W_ih, b_ih, b_hh, s_new);

    // 5) heads on z_t (uses PRE-update s_t)
    heads_kernel<<<(B_BATCH / 8), 256, 0, stream>>>(
        h_t, sent, s_t, h_ctx, W_e, b_e, W_s, b_s, out_e, out_s);

    // 6) deterministic scatter: copy image, mark winners, write winning rows
    winner_init<<<NSLOT / 256, 256, 0, stream>>>(winner);
    winner_mark<<<B_BATCH / 256, 256, 0, stream>>>(slot, winner);
    hipMemcpyAsync(out_mem, memory, (size_t)NSLOT * H_MEM * sizeof(float),
                   hipMemcpyDeviceToDevice, stream);
    scatter_rows<<<(B_BATCH * 32 + 255) / 256, 256, 0, stream>>>(slot, winner, s_new, out_mem);
}
